// EpiModule_76888504533536
// MI455X (gfx1250) — matrix operand order repaired and checked by reading, not compiled, not run on hardware
//
#include <hip/hip_runtime.h>
#include <hip/hip_bf16.h>

// ---------------------------------------------------------------------------
// Epipolar attention for MI455X (gfx1250, wave32, WMMA f32_16x16x32_f16)
// Round 2: software-pipelined (double-buffered) fragment loads so WMMAs
// overlap in-flight global loads instead of s_wait_loadcnt 0 serialization.
// ---------------------------------------------------------------------------

typedef __attribute__((ext_vector_type(16))) _Float16 v16h;
typedef __attribute__((ext_vector_type(8)))  _Float16 v8h;
typedef __attribute__((ext_vector_type(8)))  float    v8f;

#define BATCH  32
#define NSEQ   1024
#define CDIM   512
#define HEADSN 8
#define DHEAD  64

__device__ __forceinline__ int lane_id() { return (int)(threadIdx.x & 31); }

// D = A(16x32) * B(32x16) + C, f16 inputs, f32 accumulate
__device__ __forceinline__ v8f wmma_f16(v16h a, v16h b, v8f c) {
    return __builtin_amdgcn_wmma_f32_16x16x32_f16(
        /*neg_a=*/false, a, /*neg_b=*/false, b,
        /*c_mod=*/(short)0, c, /*reuse_a=*/false, /*reuse_b=*/false);
}

// A fragment, 16x32 f16 from row-major src (rows = M, cols = K), ld elems.
// Lanes 0-15 -> M=lane, K = 0..15; lanes 16-31 ->
// M=lane-16, K = 16..31.  Two contiguous 16-byte loads per lane.
__device__ __forceinline__ v16h load_frag_a(const _Float16* src, int ld) {
    const int lane = lane_id();
    const int m  = lane & 15;
    const int kb = (lane < 16) ? 0 : 16;
    v8h lo = *(const v8h*)(src + (size_t)m * ld + kb);
    v8h hi = *(const v8h*)(src + (size_t)m * ld + kb + 8);
    v16h a;
#pragma unroll
    for (int i = 0; i < 8; ++i) { a[i] = lo[i]; a[i + 8] = hi[i]; }
    return a;
}

// B fragment, 32x16 f16 from a TRANSPOSED source: srcT[col][k], ld elems.
// Lanes 0-15 -> col=lane, K=0..15; lanes 16-31 -> col=lane-16, K=16..31.
__device__ __forceinline__ v16h load_frag_bT(const _Float16* srcT, int ld) {
    const int lane = lane_id();
    const int n  = lane & 15;
    const int kb = (lane < 16) ? 0 : 16;
    v8h lo = *(const v8h*)(srcT + (size_t)n * ld + kb);
    v8h hi = *(const v8h*)(srcT + (size_t)n * ld + kb + 8);
    v16h b;
#pragma unroll
    for (int i = 0; i < 8; ++i) { b[i] = lo[i]; b[i + 8] = hi[i]; }
    return b;
}

// ---------------------------------------------------------------------------
// Pipelined 32x64 GEMM tile per wave over K = CDIM (step 32, 2 buffers).
// A: rows x CDIM row-major (rowbase applied).  Bt: cols x CDIM (colbase
// applied, transposed weight layout).  acc[ra*4+t] = row-tile ra, col-tile t.
// ---------------------------------------------------------------------------
__device__ __forceinline__ void gemm32x64(const _Float16* __restrict__ A,
                                          const _Float16* __restrict__ Bt,
                                          v8f (&acc)[8]) {
    v16h aA[2], bA[4], aB[2], bB[4];
#pragma unroll
    for (int ra = 0; ra < 2; ++ra)
        aA[ra] = load_frag_a(A + (size_t)(16 * ra) * CDIM, CDIM);
#pragma unroll
    for (int t = 0; t < 4; ++t)
        bA[t] = load_frag_bT(Bt + (size_t)(16 * t) * CDIM, CDIM);

#pragma unroll
    for (int k = 0; k < CDIM; k += 64) {
        // issue loads for k+32 into buffer B (before consuming buffer A)
#pragma unroll
        for (int ra = 0; ra < 2; ++ra)
            aB[ra] = load_frag_a(A + (size_t)(16 * ra) * CDIM + k + 32, CDIM);
#pragma unroll
        for (int t = 0; t < 4; ++t)
            bB[t] = load_frag_bT(Bt + (size_t)(16 * t) * CDIM + k + 32, CDIM);
        // compute with buffer A (k)
#pragma unroll
        for (int ra = 0; ra < 2; ++ra)
#pragma unroll
            for (int t = 0; t < 4; ++t)
                acc[ra * 4 + t] = wmma_f16(aA[ra], bA[t], acc[ra * 4 + t]);
        // issue loads for k+64 into buffer A
        if (k + 64 < CDIM) {
#pragma unroll
            for (int ra = 0; ra < 2; ++ra)
                aA[ra] = load_frag_a(A + (size_t)(16 * ra) * CDIM + k + 64, CDIM);
#pragma unroll
            for (int t = 0; t < 4; ++t)
                bA[t] = load_frag_bT(Bt + (size_t)(16 * t) * CDIM + k + 64, CDIM);
        }
        // compute with buffer B (k+32)
#pragma unroll
        for (int ra = 0; ra < 2; ++ra)
#pragma unroll
            for (int t = 0; t < 4; ++t)
                acc[ra * 4 + t] = wmma_f16(aB[ra], bB[t], acc[ra * 4 + t]);
    }
}

// ---------------------------------------------------------------------------
// 1) dtype conversion kernels
// ---------------------------------------------------------------------------
__global__ void convert_f32_to_f16(const float* __restrict__ x,
                                   _Float16* __restrict__ xh, int n) {
    for (int i = blockIdx.x * blockDim.x + threadIdx.x; i < n;
         i += gridDim.x * blockDim.x)
        xh[i] = (_Float16)x[i];
}

// Wt[c_out][c_in] = W[c_in][c_out]  (so B fragments read contiguously)
__global__ void convert_weight_T(const float* __restrict__ w,
                                 _Float16* __restrict__ wt) {
    int i = blockIdx.x * blockDim.x + threadIdx.x;   // over 512*512
    int r = i >> 9, c = i & 511;
    wt[(size_t)c * CDIM + r] = (_Float16)w[i];
}

// ---------------------------------------------------------------------------
// 2) mask precompute: per-query epipolar line coeffs + per-batch band/decay.
//    band = (3/128) * max_{n,m} |Fc_n . coord_m| / (||ab_n|| + 1e-6);
//    inner max over m (|affine|, convex) is attained at the 4 grid corners.
// ---------------------------------------------------------------------------
__global__ void prep_mask_kernel(const float* __restrict__ F_mats,
                                 float4* __restrict__ Fc4,
                                 float* __restrict__ bandArr,
                                 float* __restrict__ decayArr) {
    const int b = blockIdx.x;
    __shared__ float red[256];
    const float* F = F_mats + (size_t)b * 9;
    const float f00 = F[0], f01 = F[1], f02 = F[2];
    const float f10 = F[3], f11 = F[4], f12 = F[5];
    const float f20 = F[6], f21 = F[7], f22 = F[8];
    const bool first = (b % 16) == 0;   // FRAMES_PER_VIDEO
    float localmax = 0.f;
    for (int n = threadIdx.x; n < NSEQ; n += 256) {
        float x = 8.f * (float)(n & 31) + 3.5f;
        float y = 8.f * (float)(n >> 5) + 3.5f;
        float c0 = f00 * x + f01 * y + f02;
        float c1 = f10 * x + f11 * y + f12;
        float c2 = f20 * x + f21 * y + f22;
        if (first) { c0 = 0.f; c1 = -1.f; c2 = y; }
        float inv = 1.f / (sqrtf(c0 * c0 + c1 * c1) + 1e-6f);
        Fc4[(size_t)b * NSEQ + n] = make_float4(c0, c1, c2, inv);
        const float lo = 3.5f, hi = 251.5f;
        float m1 = fabsf(c0 * lo + c1 * lo + c2);
        float m2 = fabsf(c0 * hi + c1 * lo + c2);
        float m3 = fabsf(c0 * lo + c1 * hi + c2);
        float m4 = fabsf(c0 * hi + c1 * hi + c2);
        localmax = fmaxf(localmax, fmaxf(fmaxf(m1, m2), fmaxf(m3, m4)) * inv);
    }
    red[threadIdx.x] = localmax;
    __syncthreads();
    for (int s = 128; s > 0; s >>= 1) {
        if ((int)threadIdx.x < s)
            red[threadIdx.x] = fmaxf(red[threadIdx.x], red[threadIdx.x + s]);
        __syncthreads();
    }
    if (threadIdx.x == 0) {
        float band = (3.0f / 128.0f) * red[0];
        bandArr[b]  = band;
        decayArr[b] = 3.0f / (band + 1e-6f);
    }
}

// ---------------------------------------------------------------------------
// 3) fused QKV projection.  grid = (B, N/64, 12); 4 waves / block.
//    Each wave: 32 rows x 64 cols (block tile 64 x 128).
//    Q pre-scaled by D^-1/2; V stored transposed (D x N).
// ---------------------------------------------------------------------------
__global__ void __launch_bounds__(128)
qkv_kernel(const _Float16* __restrict__ Xh,
           const _Float16* __restrict__ WtQ, const _Float16* __restrict__ WtK,
           const _Float16* __restrict__ WtV,
           _Float16* __restrict__ Q, _Float16* __restrict__ K,
           _Float16* __restrict__ Vt) {
    const int b    = blockIdx.x;
    const int wave = threadIdx.x >> 5;
    const int rowbase = blockIdx.y * 64 + (wave & 1) * 32;
    const int z       = blockIdx.z;         // 0..11
    const int which   = z >> 2;             // 0=Q 1=K 2=V
    const int colbase = (z & 3) * 128 + (wave >> 1) * 64;
    const _Float16* Wt = (which == 0) ? WtQ : (which == 1) ? WtK : WtV;
    const _Float16* A  = Xh + ((size_t)b * NSEQ + rowbase) * CDIM;

    v8f acc[8] = {};
    gemm32x64(A, Wt + (size_t)colbase * CDIM, acc);

    const int lane = lane_id();
    const int n  = lane & 15;
    const int mb = (lane < 16) ? 0 : 8;
    const float qscale = 0.125f;            // D^-1/2
#pragma unroll
    for (int ra = 0; ra < 2; ++ra) {
#pragma unroll
        for (int t = 0; t < 4; ++t) {
            const int c = colbase + 16 * t + n;
            const int h = c >> 6, d = c & 63;
#pragma unroll
            for (int r = 0; r < 8; ++r) {
                const int q = rowbase + 16 * ra + mb + r;
                float v = acc[ra * 4 + t][r];
                if (which == 0)
                    Q[(((size_t)b * HEADSN + h) * NSEQ + q) * DHEAD + d] =
                        (_Float16)(v * qscale);
                else if (which == 1)
                    K[(((size_t)b * HEADSN + h) * NSEQ + q) * DHEAD + d] =
                        (_Float16)v;
                else
                    Vt[(((size_t)b * HEADSN + h) * DHEAD + d) * NSEQ + q] =
                        (_Float16)v;
            }
        }
    }
}

// ---------------------------------------------------------------------------
// 4) flash attention with fused epipolar mask.
//    grid = (N/128, H, B); 8 waves / block; each wave owns 16 queries.
//    K fragments double-buffered across 32-key tiles; V fragments issued at
//    the top of each step so the softmax VALU work hides their latency.
// ---------------------------------------------------------------------------
__global__ void __launch_bounds__(256)
attn_kernel(const _Float16* __restrict__ Q, const _Float16* __restrict__ K,
            const _Float16* __restrict__ Vt, const float4* __restrict__ Fc4,
            const float* __restrict__ bandArr, const float* __restrict__ decayArr,
            _Float16* __restrict__ att) {
    const int b     = blockIdx.z;
    const int h     = blockIdx.y;
    const int wave  = threadIdx.x >> 5;
    const int qbase = blockIdx.x * 128 + wave * 16;
    const int lane  = lane_id();
    const int n  = lane & 15;
    const int mb = (lane < 16) ? 0 : 8;

    __shared__ __attribute__((aligned(16))) _Float16 plds[8][16][32];

    const float band  = bandArr[b];
    const float decay = decayArr[b];
    float4 fc[8];
#pragma unroll
    for (int r = 0; r < 8; ++r)
        fc[r] = Fc4[(size_t)b * NSEQ + qbase + mb + r];

    const _Float16* Qp = Q  + (((size_t)b * HEADSN + h) * NSEQ + qbase) * DHEAD;
    const _Float16* Kp = K  + (((size_t)b * HEADSN + h) * NSEQ) * DHEAD;
    const _Float16* Vp = Vt + (((size_t)b * HEADSN + h) * DHEAD) * NSEQ;

    const v16h qa0 = load_frag_a(Qp, DHEAD);        // d = 0..31
    const v16h qa1 = load_frag_a(Qp + 32, DHEAD);   // d = 32..63

    v8f o[4] = {};
    float mrow[8], lrow[8];
#pragma unroll
    for (int r = 0; r < 8; ++r) { mrow[r] = -1e30f; lrow[r] = 0.f; }

    auto load_kf = [&](int kb, v16h (&kf)[4]) {
        kf[0] = load_frag_bT(Kp + (size_t)kb * DHEAD, DHEAD);
        kf[1] = load_frag_bT(Kp + (size_t)kb * DHEAD + 32, DHEAD);
        kf[2] = load_frag_bT(Kp + (size_t)(kb + 16) * DHEAD, DHEAD);
        kf[3] = load_frag_bT(Kp + (size_t)(kb + 16) * DHEAD + 32, DHEAD);
    };

    auto process = [&](int kb, v16h (&kf)[4]) {
        // issue V loads first: latency hidden behind score WMMAs + softmax
        v16h vf[4];
#pragma unroll
        for (int t = 0; t < 4; ++t)
            vf[t] = load_frag_bT(Vp + (size_t)(16 * t) * NSEQ + kb, NSEQ);
        // scores: two 16q x 16k tiles, K-dim 64 = 2 x 32
        v8f s0 = {}, s1 = {};
        s0 = wmma_f16(qa0, kf[0], s0);
        s0 = wmma_f16(qa1, kf[1], s0);
        s1 = wmma_f16(qa0, kf[2], s1);
        s1 = wmma_f16(qa1, kf[3], s1);
        // epipolar mask on the fly
        {
            const int m0 = kb + n, m1 = kb + 16 + n;
            const float x0 = 8.f * (float)(m0 & 31) + 3.5f;
            const float y0 = 8.f * (float)(m0 >> 5) + 3.5f;
            const float x1 = 8.f * (float)(m1 & 31) + 3.5f;
            const float y1 = 8.f * (float)(m1 >> 5) + 3.5f;
#pragma unroll
            for (int r = 0; r < 8; ++r) {
                float c0 = fabsf(fc[r].x * x0 + fc[r].y * y0 + fc[r].z) * fc[r].w;
                float c1 = fabsf(fc[r].x * x1 + fc[r].y * y1 + fc[r].z) * fc[r].w;
                s0[r] -= fmaxf(c0 - band, 0.f) * decay;
                s1[r] -= fmaxf(c1 - band, 0.f) * decay;
            }
        }
        // online softmax; rows 0-7 in lanes 0-15, rows 8-15 in lanes 16-31
        float scale[8];
#pragma unroll
        for (int r = 0; r < 8; ++r) {
            float t = fmaxf(s0[r], s1[r]);
            t = fmaxf(t, __shfl_xor(t, 1, 32));
            t = fmaxf(t, __shfl_xor(t, 2, 32));
            t = fmaxf(t, __shfl_xor(t, 4, 32));
            t = fmaxf(t, __shfl_xor(t, 8, 32));
            const float mnew = fmaxf(mrow[r], t);
            scale[r] = __expf(mrow[r] - mnew);
            mrow[r]  = mnew;
            const float p0 = __expf(s0[r] - mnew);
            const float p1 = __expf(s1[r] - mnew);
            s0[r] = p0; s1[r] = p1;
            float sum = p0 + p1;
            sum += __shfl_xor(sum, 1, 32);
            sum += __shfl_xor(sum, 2, 32);
            sum += __shfl_xor(sum, 4, 32);
            sum += __shfl_xor(sum, 8, 32);
            lrow[r] = lrow[r] * scale[r] + sum;
        }
#pragma unroll
        for (int t = 0; t < 4; ++t)
#pragma unroll
            for (int r = 0; r < 8; ++r) o[t][r] *= scale[r];
        // P (C layout) -> LDS -> A fragment (same-wave DS ordering)
#pragma unroll
        for (int r = 0; r < 8; ++r) {
            plds[wave][mb + r][n]      = (_Float16)s0[r];
            plds[wave][mb + r][16 + n] = (_Float16)s1[r];
        }
        const v16h pa = load_frag_a(&plds[wave][0][0], 32);
#pragma unroll
        for (int t = 0; t < 4; ++t) o[t] = wmma_f16(pa, vf[t], o[t]);
    };

    v16h kfA[4], kfB[4];
    load_kf(0, kfA);
    for (int kb = 0; kb < NSEQ; kb += 64) {
        load_kf(kb + 32, kfB);       // next-tile K loads in flight
        process(kb, kfA);
        if (kb + 64 < NSEQ) load_kf(kb + 64, kfA);
        process(kb + 32, kfB);
    }

    // normalize and write attention output (f16, (B,N,C) layout)
#pragma unroll
    for (int r = 0; r < 8; ++r) {
        const float inv = 1.f / lrow[r];
        const size_t row = (size_t)b * NSEQ + qbase + mb + r;
#pragma unroll
        for (int t = 0; t < 4; ++t)
            att[row * CDIM + h * DHEAD + 16 * t + n] = (_Float16)(o[t][r] * inv);
    }
}

// ---------------------------------------------------------------------------
// 5) output projection: out = att @ Wo  (f32 output)
//    grid = (B*N/64, C/128); 4 waves / block, 32 rows x 64 cols per wave.
// ---------------------------------------------------------------------------
__global__ void __launch_bounds__(128)
proj_kernel(const _Float16* __restrict__ att, const _Float16* __restrict__ WtO,
            float* __restrict__ out) {
    const int wave = threadIdx.x >> 5;
    const int rowbase = blockIdx.x * 64 + (wave & 1) * 32;
    const int colbase = blockIdx.y * 128 + (wave >> 1) * 64;

    v8f acc[8] = {};
    gemm32x64(att + (size_t)rowbase * CDIM,
              WtO + (size_t)colbase * CDIM, acc);

    const int lane = lane_id();
    const int n  = lane & 15;
    const int mb = (lane < 16) ? 0 : 8;
#pragma unroll
    for (int ra = 0; ra < 2; ++ra)
#pragma unroll
        for (int t = 0; t < 4; ++t)
#pragma unroll
            for (int r = 0; r < 8; ++r)
                out[(size_t)(rowbase + 16 * ra + mb + r) * CDIM + colbase +
                    16 * t + n] = acc[ra * 4 + t][r];
}

// ---------------------------------------------------------------------------
// host orchestration
// ---------------------------------------------------------------------------
extern "C" void kernel_launch(void* const* d_in, const int* in_sizes, int n_in,
                              void* d_out, int out_size, void* d_ws, size_t ws_size,
                              hipStream_t stream) {
    (void)in_sizes; (void)n_in; (void)out_size; (void)ws_size;
    const float* hidden = (const float*)d_in[0];
    const float* F_mats = (const float*)d_in[1];
    const float* Wq     = (const float*)d_in[2];
    const float* Wk     = (const float*)d_in[3];
    const float* Wv     = (const float*)d_in[4];
    const float* Wo     = (const float*)d_in[5];
    float* out          = (float*)d_out;

    char* ws = (char*)d_ws;
    size_t off = 0;
    auto alloc = [&](size_t bytes) -> char* {
        char* p = ws + off;
        off = (off + bytes + 255) & ~(size_t)255;
        return p;
    };
    const size_t BNC = (size_t)BATCH * NSEQ * CDIM;     // 16,777,216
    _Float16* Xh  = (_Float16*)alloc(BNC * 2);
    _Float16* WtQ = (_Float16*)alloc((size_t)CDIM * CDIM * 2);
    _Float16* WtK = (_Float16*)alloc((size_t)CDIM * CDIM * 2);
    _Float16* WtV = (_Float16*)alloc((size_t)CDIM * CDIM * 2);
    _Float16* WtO = (_Float16*)alloc((size_t)CDIM * CDIM * 2);
    _Float16* Qh  = (_Float16*)alloc(BNC * 2);
    _Float16* Kh  = (_Float16*)alloc(BNC * 2);
    _Float16* Vth = (_Float16*)alloc(BNC * 2);
    _Float16* Att = (_Float16*)alloc(BNC * 2);
    float4*   Fc4 = (float4*)  alloc((size_t)BATCH * NSEQ * sizeof(float4));
    float*    Bnd = (float*)   alloc(BATCH * sizeof(float));
    float*    Dcy = (float*)   alloc(BATCH * sizeof(float));

    // 1) conversions
    convert_f32_to_f16<<<2048, 256, 0, stream>>>(hidden, Xh, (int)BNC);
    convert_weight_T<<<(CDIM * CDIM) / 256, 256, 0, stream>>>(Wq, WtQ);
    convert_weight_T<<<(CDIM * CDIM) / 256, 256, 0, stream>>>(Wk, WtK);
    convert_weight_T<<<(CDIM * CDIM) / 256, 256, 0, stream>>>(Wv, WtV);
    convert_weight_T<<<(CDIM * CDIM) / 256, 256, 0, stream>>>(Wo, WtO);

    // 2) mask precompute
    prep_mask_kernel<<<BATCH, 256, 0, stream>>>(F_mats, Fc4, Bnd, Dcy);

    // 3) QKV projection
    qkv_kernel<<<dim3(BATCH, NSEQ / 64, 12), 128, 0, stream>>>(
        Xh, WtQ, WtK, WtV, Qh, Kh, Vth);

    // 4) flash attention + fused epipolar mask
    attn_kernel<<<dim3(NSEQ / 128, HEADSN, BATCH), 256, 0, stream>>>(
        Qh, Kh, Vth, Fc4, Bnd, Dcy, Att);

    // 5) output projection
    proj_kernel<<<dim3((BATCH * NSEQ) / 64, CDIM / 128), 128, 0, stream>>>(
        Att, WtO, out);
}
